// InstabilityGatedAttention_52158082842707
// MI455X (gfx1250) — matrix operand order repaired and checked by reading, not compiled, not run on hardware
//
#include <hip/hip_runtime.h>
#include <hip/hip_bf16.h>

typedef __attribute__((ext_vector_type(16))) _Float16 v16h;
typedef __attribute__((ext_vector_type(8)))  _Float16 v8h;
typedef __attribute__((ext_vector_type(4)))  _Float16 v4h;
typedef __attribute__((ext_vector_type(8)))  float    v8f;
typedef __attribute__((ext_vector_type(4)))  unsigned int u32x4;
typedef __attribute__((ext_vector_type(8)))  int          i32x8;
typedef __attribute__((ext_vector_type(4)))  int          i32x4;

#define BATCH 2
#define SEQ   2048
#define EMB   1024
#define NHEAD 16
#define DHEAD 64
#define HHID  512          // probe hidden
#define MTOT  (BATCH*SEQ)  // 4096 tokens

#if __has_builtin(__builtin_amdgcn_tensor_load_to_lds)
#define HAVE_TDM 1
#else
#define HAVE_TDM 0
#endif

__device__ __forceinline__ float gelu_exact(float x) {
    return 0.5f * x * (1.0f + erff(x * 0.70710678118654752f));
}

// Low 32 bits of a flat pointer to LDS == DS-space byte offset (aperture mapping).
__device__ __forceinline__ unsigned int lds_off(const void* p) {
    return (unsigned int)(unsigned long long)(uintptr_t)p;
}

#if HAVE_TDM
// Issue a TDM 2-D tile load: tile_d0 (contiguous, elems) x tile_d1 lines,
// line stride = stride_d0 elems, f16 data, into LDS at lds_byte.
__device__ __forceinline__ void tdm_load_tile_f16(
    unsigned int lds_byte, const void* gptr,
    unsigned int tile_d0, unsigned int tile_d1, unsigned int stride_d0)
{
    unsigned long long ga = (unsigned long long)(uintptr_t)gptr;
    u32x4 g0;
    g0[0] = 1u;                                   // count=1, user descriptor
    g0[1] = lds_byte;                             // lds_addr
    g0[2] = (unsigned int)ga;                     // global_addr[31:0]
    g0[3] = (unsigned int)((ga >> 32) & 0x1FFFFFFu) | (2u << 30);  // addr[56:32], type=2
    i32x8 g1;
    g1[0] = (int)(1u << 16);                      // wg_mask=0, data_size=1 (2 bytes)
    g1[1] = (int)((tile_d0 & 0xFFFFu) << 16);     // tensor_dim0[15:0] = tile_d0
    g1[2] = (int)((tile_d1 & 0xFFFFu) << 16);     // tensor_dim0 hi=0, tensor_dim1 lo
    g1[3] = (int)((tile_d0 & 0xFFFFu) << 16);     // tensor_dim1 hi=0, tile_dim0
    g1[4] = (int)(tile_d1 & 0xFFFFu);             // tile_dim1, tile_dim2=0
    g1[5] = (int)stride_d0;                       // tensor_dim0_stride[31:0]
    g1[6] = 0;                                    // stride hi, dim1_stride lo = 0
    g1[7] = 0;
    i32x4 gz = {0, 0, 0, 0};
#if defined(__clang_major__) && (__clang_major__ >= 23)
    i32x8 gz8 = {0, 0, 0, 0, 0, 0, 0, 0};
    __builtin_amdgcn_tensor_load_to_lds(g0, g1, gz, gz, gz8, 0);
#else
    __builtin_amdgcn_tensor_load_to_lds(g0, g1, gz, gz, 0);
#endif
}
#endif

// ---------------------------------------------------------------------------
// f32 -> f16 elementwise convert (vectorized x4)
// ---------------------------------------------------------------------------
__global__ __launch_bounds__(256) void cvt_f16_kernel(
    const float* __restrict__ in, _Float16* __restrict__ out, int n4)
{
    int i = blockIdx.x * 256 + threadIdx.x;
    if (i < n4) {
        const float4 v = ((const float4*)in)[i];
        v4h o = { (_Float16)v.x, (_Float16)v.y, (_Float16)v.z, (_Float16)v.w };
        ((v4h*)out)[i] = o;
    }
}

// ---------------------------------------------------------------------------
// Transpose + convert: in f32 [K][N] -> out f16 [N][K]   (32x32 LDS tiles)
// ---------------------------------------------------------------------------
__global__ __launch_bounds__(256) void transpose_cvt_kernel(
    const float* __restrict__ in, _Float16* __restrict__ outT, int K, int N)
{
    __shared__ float tile[32][33];
    const int n0 = blockIdx.x * 32, k0 = blockIdx.y * 32;
    const int tx = threadIdx.x & 31, ty = threadIdx.x >> 5;   // 8 rows at a time
    #pragma unroll
    for (int i = 0; i < 32; i += 8)
        tile[ty + i][tx] = in[(size_t)(k0 + ty + i) * N + n0 + tx];
    __syncthreads();
    #pragma unroll
    for (int i = 0; i < 32; i += 8)
        outT[(size_t)(n0 + ty + i) * K + k0 + tx] = (_Float16)tile[tx][ty + i];
}

// ---------------------------------------------------------------------------
// Tiled WMMA GEMM:  C[M x N] = A[M x K](f16) * WT[N x K](f16)^T + bias
// Block = 128 threads (4 waves), block tile 64x64, k-step 32.
// TDM double-buffered staging (wave 0 drives the Tensor Data Mover).
// Epilogue MODE selects destination layout.
// ---------------------------------------------------------------------------
enum { M_ROWF32 = 0, M_QFRAG = 1, M_KFRAG = 2, M_VFRAG = 3 };

template <int MODE>
__global__ __launch_bounds__(128) void gemm_wmma_kernel(
    const _Float16* __restrict__ A, const _Float16* __restrict__ WT,
    const float* __restrict__ bias,
    float* __restrict__ outF, _Float16* __restrict__ outH,
    int Kdim, int Ndim, float outScale)
{
    __shared__ __align__(32) _Float16 lA[2][64 * 32];   // [row 64][k 32]
    __shared__ __align__(32) _Float16 lW[2][64 * 32];   // [col 64][k 32]

    const int tid  = threadIdx.x;
    const int lane = tid & 31;
    const int wave = tid >> 5;
    const int nbase = blockIdx.x * 64;
    const int mbase = blockIdx.y * 64;

    const _Float16* Abase = A  + (size_t)mbase * Kdim;
    const _Float16* Wbase = WT + (size_t)nbase * Kdim;

    v8f acc[4] = {v8f{}, v8f{}, v8f{}, v8f{}};
    const int nk = Kdim >> 5;

    // Prologue: stage k-tile 0 into buffer 0
#if HAVE_TDM
    if (wave == 0) {
        tdm_load_tile_f16(lds_off(&lA[0][0]), Abase, 32, 64, Kdim);
        tdm_load_tile_f16(lds_off(&lW[0][0]), Wbase, 32, 64, Kdim);
        __builtin_amdgcn_s_wait_tensorcnt(0);
    }
#else
    *(v16h*)(&lA[0][tid * 16]) = *(const v16h*)(Abase + (size_t)(tid >> 1) * Kdim + (tid & 1) * 16);
    *(v16h*)(&lW[0][tid * 16]) = *(const v16h*)(Wbase + (size_t)(tid >> 1) * Kdim + (tid & 1) * 16);
#endif
    __syncthreads();

    for (int kt = 0; kt < nk; ++kt) {
        const int cur = kt & 1, nxt = cur ^ 1;
        // Prefetch next k-tile into the other buffer (overlaps with WMMA below)
        if (kt + 1 < nk) {
#if HAVE_TDM
            if (wave == 0) {
                tdm_load_tile_f16(lds_off(&lA[nxt][0]), Abase + (kt + 1) * 32, 32, 64, Kdim);
                tdm_load_tile_f16(lds_off(&lW[nxt][0]), Wbase + (kt + 1) * 32, 32, 64, Kdim);
            }
#else
            *(v16h*)(&lA[nxt][tid * 16]) =
                *(const v16h*)(Abase + (size_t)(tid >> 1) * Kdim + (kt + 1) * 32 + (tid & 1) * 16);
            *(v16h*)(&lW[nxt][tid * 16]) =
                *(const v16h*)(Wbase + (size_t)(tid >> 1) * Kdim + (kt + 1) * 32 + (tid & 1) * 16);
#endif
        }

        // A fragment (16x32, contiguous k = 16*g + s per lane half, same as B below)
        const int g = lane >> 4;
        const _Float16* ap = &lA[cur][(wave * 16 + (lane & 15)) * 32];
        v16h aF = *(const v16h*)(ap + g * 16);

        #pragma unroll
        for (int n = 0; n < 4; ++n) {
            const _Float16* wp = &lW[cur][(n * 16 + (lane & 15)) * 32 + g * 16];
            v16h bF = *(const v16h*)wp;
            acc[n] = __builtin_amdgcn_wmma_f32_16x16x32_f16(
                false, aF, false, bF, (short)0, acc[n], false, false);
        }

#if HAVE_TDM
        if (wave == 0) __builtin_amdgcn_s_wait_tensorcnt(0);
#endif
        __syncthreads();
    }

    // Epilogue. C-layout: lanes 0-15 -> M=r, lanes 16-31 -> M=8+r; N = lane%16.
    const int g16 = lane >> 4;
    const int nn  = lane & 15;
    #pragma unroll
    for (int n = 0; n < 4; ++n) {
        int col = nbase + n * 16 + nn;
        float bv = bias[col];
        #pragma unroll
        for (int r = 0; r < 8; ++r) {
            int m = mbase + wave * 16 + g16 * 8 + r;
            float val = (acc[n][r] + bv) * outScale;
            if constexpr (MODE == M_ROWF32) {
                outF[(size_t)m * Ndim + col] = val;
            } else {
                int b  = m >> 11;         // token / SEQ
                int s  = m & (SEQ - 1);
                int hh = col >> 6;        // head
                int d  = col & 63;
                _Float16 hv = (_Float16)val;
                if constexpr (MODE == M_QFRAG) {
                    int qt = s >> 4, Mq = s & 15;
                    int c = d >> 5, dc = d & 31;
                    int gg = dc >> 4;
                    int j  = dc & 15;
                    size_t idx = ((((size_t)(b * NHEAD + hh) * (SEQ / 16) + qt) * 2 + c) * 32
                                  + (gg * 16 + Mq)) * 16 + j;
                    outH[idx] = hv;
                } else if constexpr (MODE == M_KFRAG) {
                    int kt = s >> 4, kn = s & 15;
                    int c = d >> 5, dc = d & 31;
                    int lg = dc >> 4, j = dc & 15;
                    size_t idx = ((((size_t)(b * NHEAD + hh) * (SEQ / 16) + kt) * 2 + c) * 32
                                  + (lg * 16 + kn)) * 16 + j;
                    outH[idx] = hv;
                } else { // M_VFRAG
                    int t32 = s >> 5, k32 = s & 31;
                    int lg = (k32 >> 3) & 1, j = (k32 & 7) + ((k32 >> 4) << 3);
                    int nq = d >> 4, dn = d & 15;
                    size_t idx = ((((size_t)(b * NHEAD + hh) * (SEQ / 32) + t32) * 4 + nq) * 32
                                  + (lg * 16 + dn)) * 16 + j;
                    outH[idx] = hv;
                }
            }
        }
    }
}

// ---------------------------------------------------------------------------
// Drift probe: h -> LayerNorm -> GELU -> dot(dp_w2) -> gate MLP -> tanh
// ---------------------------------------------------------------------------
__global__ __launch_bounds__(128) void drift_gate_kernel(
    const float* __restrict__ hbuf,
    const float* __restrict__ ln_g, const float* __restrict__ ln_b,
    const float* __restrict__ w2,  const float* __restrict__ b2,
    const float* __restrict__ gw1, const float* __restrict__ gb1,
    const float* __restrict__ gw2, const float* __restrict__ gb2,
    float* __restrict__ gate)
{
    __shared__ float red[128];
    const int row = blockIdx.x;
    const int t   = threadIdx.x;
    const float* hr = hbuf + (size_t)row * HHID;

    float v[4], s = 0.f, ss = 0.f;
    #pragma unroll
    for (int i = 0; i < 4; ++i) {
        v[i] = hr[t + i * 128];
        s += v[i]; ss += v[i] * v[i];
    }
    red[t] = s; __syncthreads();
    for (int off = 64; off > 0; off >>= 1) { if (t < off) red[t] += red[t + off]; __syncthreads(); }
    float mean = red[0] * (1.0f / HHID);
    __syncthreads();
    red[t] = ss; __syncthreads();
    for (int off = 64; off > 0; off >>= 1) { if (t < off) red[t] += red[t + off]; __syncthreads(); }
    float var = red[0] * (1.0f / HHID) - mean * mean;
    __syncthreads();
    float rstd = rsqrtf(var + 1e-5f);

    float dot = 0.f;
    #pragma unroll
    for (int i = 0; i < 4; ++i) {
        int idx = t + i * 128;
        float x = (v[i] - mean) * rstd * ln_g[idx] + ln_b[idx];
        dot += gelu_exact(x) * w2[idx];
    }
    red[t] = dot; __syncthreads();
    for (int off = 64; off > 0; off >>= 1) { if (t < off) red[t] += red[t + off]; __syncthreads(); }
    if (t == 0) {
        float drift = red[0] + b2[0];
        float a = 0.f;
        #pragma unroll
        for (int j = 0; j < 16; ++j)
            a += gelu_exact(drift * gw1[j] + gb1[j]) * gw2[j];
        gate[row] = tanhf(a + gb2[0]);
    }
}

// ---------------------------------------------------------------------------
// Flash attention with post-softmax key gate.
// ---------------------------------------------------------------------------
__global__ __launch_bounds__(128) void flash_gated_attn_kernel(
    const _Float16* __restrict__ Qf, const _Float16* __restrict__ Kf,
    const _Float16* __restrict__ Vf, const float* __restrict__ gate,
    _Float16* __restrict__ O)
{
    __shared__ __align__(32) _Float16 lP[4][16 * 32];   // per-wave P patch

    const int lane = threadIdx.x & 31;
    const int wave = threadIdx.x >> 5;
    const int qtile = blockIdx.x * 4 + wave;        // 0..127
    const int bh    = blockIdx.y;                   // 0..31
    const int b     = bh >> 4;
    const int qbase = qtile << 4;

    const _Float16* qp = Qf + ((size_t)bh * (SEQ / 16) + qtile) * 1024;
    v16h qA0 = *(const v16h*)(qp + lane * 16);
    v16h qA1 = *(const v16h*)(qp + 512 + lane * 16);

    float rowm[8], rowl[8];
    #pragma unroll
    for (int r = 0; r < 8; ++r) { rowm[r] = -1e30f; rowl[r] = 0.f; }
    v8f acc[4] = {v8f{}, v8f{}, v8f{}, v8f{}};

    const int nkt = (qtile >> 1) + 1;
    for (int kt = 0; kt < nkt; ++kt) {
        const int kbase = kt << 5;
        const bool diag = (kt == nkt - 1);

        const _Float16* kp = Kf + ((size_t)bh * (SEQ / 16) + kt * 2) * 1024;
        v16h kB00 = *(const v16h*)(kp + lane * 16);
        v16h kB01 = *(const v16h*)(kp + 512 + lane * 16);
        v16h kB10 = *(const v16h*)(kp + 1024 + lane * 16);
        v16h kB11 = *(const v16h*)(kp + 1536 + lane * 16);

        v8f c0 = {}, c1 = {};
        c0 = __builtin_amdgcn_wmma_f32_16x16x32_f16(false, qA0, false, kB00, (short)0, c0, false, false);
        c0 = __builtin_amdgcn_wmma_f32_16x16x32_f16(false, qA1, false, kB01, (short)0, c0, false, false);
        c1 = __builtin_amdgcn_wmma_f32_16x16x32_f16(false, qA0, false, kB10, (short)0, c1, false, false);
        c1 = __builtin_amdgcn_wmma_f32_16x16x32_f16(false, qA1, false, kB11, (short)0, c1, false, false);

        const float g0 = gate[b * SEQ + kbase + (lane & 15)];
        const float g1 = gate[b * SEQ + kbase + 16 + (lane & 15)];
        const int key0 = kbase + (lane & 15);

        #pragma unroll
        for (int r = 0; r < 8; ++r) {
            int q = qbase + ((lane >> 4) << 3) + r;
            float l0 = c0[r], l1 = c1[r];
            if (diag) {
                if (key0 > q)      l0 = -1e30f;
                if (key0 + 16 > q) l1 = -1e30f;
            }
            float t = fmaxf(l0, l1);
            t = fmaxf(t, __shfl_xor(t, 1, 32));
            t = fmaxf(t, __shfl_xor(t, 2, 32));
            t = fmaxf(t, __shfl_xor(t, 4, 32));
            t = fmaxf(t, __shfl_xor(t, 8, 32));
            float mnew = fmaxf(rowm[r], t);
            float corr = __expf(rowm[r] - mnew);
            float e0 = __expf(l0 - mnew);
            float e1 = __expf(l1 - mnew);
            float se = e0 + e1;
            se += __shfl_xor(se, 1, 32);
            se += __shfl_xor(se, 2, 32);
            se += __shfl_xor(se, 4, 32);
            se += __shfl_xor(se, 8, 32);
            rowl[r] = rowl[r] * corr + se;   // denominator: ungated
            rowm[r] = mnew;
            acc[0][r] *= corr; acc[1][r] *= corr;
            acc[2][r] *= corr; acc[3][r] *= corr;
            int M = ((lane >> 4) << 3) + r;
            lP[wave][M * 32 + (lane & 15)]      = (_Float16)(e0 * g0);
            lP[wave][M * 32 + 16 + (lane & 15)] = (_Float16)(e1 * g1);
        }
        asm volatile("s_wait_dscnt 0x0" ::: "memory");   // order LDS RAW within wave

        const _Float16* pp = &lP[wave][0];
        const int prow = lane & 15, pg = lane >> 4;
        v8h plo = *(const v8h*)(pp + prow * 32 + pg * 8);
        v8h phi = *(const v8h*)(pp + prow * 32 + 16 + pg * 8);
        v16h pA = __builtin_shufflevector(plo, phi, 0,1,2,3,4,5,6,7,8,9,10,11,12,13,14,15);

        const _Float16* vp = Vf + ((size_t)bh * (SEQ / 32) + kt) * 2048;
        #pragma unroll
        for (int n = 0; n < 4; ++n) {
            v16h vB = *(const v16h*)(vp + n * 512 + lane * 16);
            acc[n] = __builtin_amdgcn_wmma_f32_16x16x32_f16(
                false, pA, false, vB, (short)0, acc[n], false, false);
        }
    }

    #pragma unroll
    for (int n = 0; n < 4; ++n) {
        int col = (bh & 15) * DHEAD + n * 16 + (lane & 15);
        #pragma unroll
        for (int r = 0; r < 8; ++r) {
            int row = b * SEQ + qbase + ((lane >> 4) << 3) + r;
            O[(size_t)row * EMB + col] = (_Float16)(acc[n][r] / rowl[r]);
        }
    }
}

// ---------------------------------------------------------------------------
extern "C" void kernel_launch(void* const* d_in, const int* in_sizes, int n_in,
                              void* d_out, int out_size, void* d_ws, size_t ws_size,
                              hipStream_t stream) {
    const float* x    = (const float*)d_in[0];
    const float* Wq   = (const float*)d_in[1];
    const float* bq   = (const float*)d_in[2];
    const float* Wk   = (const float*)d_in[3];
    const float* bk   = (const float*)d_in[4];
    const float* Wv   = (const float*)d_in[5];
    const float* bv   = (const float*)d_in[6];
    const float* Wo   = (const float*)d_in[7];
    const float* bo   = (const float*)d_in[8];
    const float* dpw1 = (const float*)d_in[9];
    const float* dpb1 = (const float*)d_in[10];
    const float* lng  = (const float*)d_in[11];
    const float* lnb  = (const float*)d_in[12];
    const float* dpw2 = (const float*)d_in[13];
    const float* dpb2 = (const float*)d_in[14];
    const float* gw1  = (const float*)d_in[15];
    const float* gb1  = (const float*)d_in[16];
    const float* gw2  = (const float*)d_in[17];
    const float* gb2  = (const float*)d_in[18];
    float* out = (float*)d_out;

    char* ws = (char*)d_ws;
    const size_t MB = 1048576;
    _Float16* Qf   = (_Float16*)(ws);              // 8 MB
    _Float16* Kf   = (_Float16*)(ws + 8  * MB);    // 8 MB
    _Float16* Vf   = (_Float16*)(ws + 16 * MB);    // 8 MB
    _Float16* Of   = (_Float16*)(ws + 24 * MB);    // 8 MB
    float*    hb   = (float*)  (ws + 32 * MB);     // 8 MB
    float*    gt   = (float*)  (ws + 40 * MB);     // 16 KB
    _Float16* xh   = (_Float16*)(ws + 41 * MB);    // 8 MB  x in f16
    _Float16* WqT  = (_Float16*)(ws + 49 * MB);    // 2 MB  [N][K] f16
    _Float16* WkT  = (_Float16*)(ws + 51 * MB);
    _Float16* WvT  = (_Float16*)(ws + 53 * MB);
    _Float16* WoT  = (_Float16*)(ws + 55 * MB);
    _Float16* w1T  = (_Float16*)(ws + 57 * MB);    // 1 MB  [512][1024]

    const float scale = 0.125f;   // DHEAD^-0.5
    dim3 blk(128);
    dim3 gE(EMB / 64, MTOT / 64);    // 16 x 64
    dim3 gH(HHID / 64, MTOT / 64);   // 8 x 64

    // Stage 0: one-shot precision/layout prep (f16 operands, weights pre-transposed)
    cvt_f16_kernel<<<dim3((MTOT * EMB / 4 + 255) / 256), dim3(256), 0, stream>>>(x, xh, MTOT * EMB / 4);
    transpose_cvt_kernel<<<dim3(EMB / 32, EMB / 32),  dim3(256), 0, stream>>>(Wq,   WqT, EMB, EMB);
    transpose_cvt_kernel<<<dim3(EMB / 32, EMB / 32),  dim3(256), 0, stream>>>(Wk,   WkT, EMB, EMB);
    transpose_cvt_kernel<<<dim3(EMB / 32, EMB / 32),  dim3(256), 0, stream>>>(Wv,   WvT, EMB, EMB);
    transpose_cvt_kernel<<<dim3(EMB / 32, EMB / 32),  dim3(256), 0, stream>>>(Wo,   WoT, EMB, EMB);
    transpose_cvt_kernel<<<dim3(HHID / 32, EMB / 32), dim3(256), 0, stream>>>(dpw1, w1T, EMB, HHID);

    // Stage 1: projections (TDM-staged WMMA GEMMs) into attention fragment layouts
    gemm_wmma_kernel<M_QFRAG><<<gE, blk, 0, stream>>>(xh, WqT, bq, nullptr, Qf, EMB, EMB, scale);
    gemm_wmma_kernel<M_KFRAG><<<gE, blk, 0, stream>>>(xh, WkT, bk, nullptr, Kf, EMB, EMB, 1.0f);
    gemm_wmma_kernel<M_VFRAG><<<gE, blk, 0, stream>>>(xh, WvT, bv, nullptr, Vf, EMB, EMB, 1.0f);
    gemm_wmma_kernel<M_ROWF32><<<gH, blk, 0, stream>>>(xh, w1T, dpb1, hb, nullptr, EMB, HHID, 1.0f);

    // Stage 2: drift probe -> per-key gate
    drift_gate_kernel<<<dim3(MTOT), blk, 0, stream>>>(hb, lng, lnb, dpw2, dpb2, gw1, gb1, gw2, gb2, gt);

    // Stage 3: gated flash attention
    flash_gated_attn_kernel<<<dim3(SEQ / 16 / 4, BATCH * NHEAD), blk, 0, stream>>>(Qf, Kf, Vf, gt, Of);

    // Stage 4: output projection
    gemm_wmma_kernel<M_ROWF32><<<gE, blk, 0, stream>>>(Of, WoT, bo, out, nullptr, EMB, EMB, 1.0f);
}
